// LSTMTagger_71871982731463
// MI455X (gfx1250) — matrix operand order repaired and checked by reading, not compiled, not run on hardware
//
#include <hip/hip_runtime.h>

// ---------------- problem dims ----------------
#define T_LEN 8192
#define F_DIM 1024
#define H_DIM 2048
#define G_DIM (4 * H_DIM)   // 8192 gate rows
#define NB_SCAN 32          // persistent blocks in the scan kernel

typedef __attribute__((ext_vector_type(16))) __bf16          v16bf;
typedef __attribute__((ext_vector_type(16))) unsigned short  v16u;
typedef __attribute__((ext_vector_type(8)))  float           v8f;

// ---------------- bf16 helpers ----------------
__device__ __forceinline__ unsigned short f2bf(float f) {
  unsigned u = __builtin_bit_cast(unsigned, f);
  unsigned r = 0x7FFFu + ((u >> 16) & 1u);          // round-to-nearest-even
  return (unsigned short)((u + r) >> 16);
}
__device__ __forceinline__ float bf_lo(unsigned w) {
  return __builtin_bit_cast(float, w << 16);
}
__device__ __forceinline__ float bf_hi(unsigned w) {
  return __builtin_bit_cast(float, w & 0xFFFF0000u);
}
__device__ __forceinline__ float sigmoidf(float x) {
  return 1.0f / (1.0f + __expf(-x));
}

// ---------------- small prep kernels ----------------
__global__ void cvt_bf16_kernel(const float* __restrict__ src,
                                unsigned short* __restrict__ dst, int n) {
  int i = blockIdx.x * blockDim.x + threadIdx.x;
  int stride = gridDim.x * blockDim.x;
  for (; i < n; i += stride) dst[i] = f2bf(src[i]);
}

__global__ void bias_sum_kernel(const float* __restrict__ a,
                                const float* __restrict__ b,
                                float* __restrict__ o, int n) {
  int i = blockIdx.x * blockDim.x + threadIdx.x;
  if (i < n) o[i] = a[i] + b[i];
}

__global__ void init_sync_kernel(unsigned* s) {
  if (threadIdx.x < 16) s[threadIdx.x] = 0u;
}

// ---------------- x_gates GEMM: [T,F] x [F,G] via bf16 WMMA ----------------
// Block = 256 threads (8 waves). Block tile: 128 (M) x 128 (N).
// Wave w owns M rows [mWave, mWave+16) and eight 16x16 N sub-tiles.
__global__ __launch_bounds__(256) void xgates_wmma_kernel(
    const unsigned short* __restrict__ Abf,   // input bf16 [T, F]
    const unsigned short* __restrict__ Wih,   // W_ih bf16 [G, F] (B = Wih^T)
    const float* __restrict__ bias,           // [G] = b_ih + b_hh
    float* __restrict__ xg)                   // [T, G] fp32
{
  const int lane   = threadIdx.x & 31;
  const int wave   = threadIdx.x >> 5;
  const int mWave  = blockIdx.y * 128 + wave * 16;
  const int nBase  = blockIdx.x * 128;
  const int lan15  = lane & 15;
  const int hiHalf = lane >> 4;                 // 0: lanes 0-15, 1: lanes 16-31
  const int kHalfA = hiHalf ? 8  : 0;           // 16-bit A layout K offset
  const int kHalfB = hiHalf ? 8  : 0;           // 16-bit B layout K offset

  v8f acc[8];
#pragma unroll
  for (int i = 0; i < 8; ++i)
#pragma unroll
    for (int e = 0; e < 8; ++e) acc[i][e] = 0.0f;

  const unsigned short* arow = Abf + (size_t)(mWave + lan15) * F_DIM;

  for (int kb = 0; kb < F_DIM; kb += 32) {
    // A fragment: 16x32 bf16, per ISA 16-bit A layout
    v16u au;
#pragma unroll
    for (int v = 0; v < 8; ++v) {
      int k0 = (v < 4 ? 2 * v : 16 + 2 * (v - 4)) + kHalfA;
      unsigned w = *(const unsigned*)(arow + kb + k0);
      au[2 * v]     = (unsigned short)(w & 0xFFFFu);
      au[2 * v + 1] = (unsigned short)(w >> 16);
    }
    v16bf a = __builtin_bit_cast(v16bf, au);

#pragma unroll
    for (int nt = 0; nt < 8; ++nt) {
      // B fragment: 32x16 bf16; column n = row n of W_ih (contiguous in k)
      const unsigned short* brow =
          Wih + (size_t)(nBase + nt * 16 + lan15) * F_DIM + kb + kHalfB;
      v16u bu;
#pragma unroll
      for (int v = 0; v < 8; ++v) {
        unsigned w = *(const unsigned*)(brow + (v < 4 ? 2 * v : 16 + 2 * (v - 4)));
        bu[2 * v]     = (unsigned short)(w & 0xFFFFu);
        bu[2 * v + 1] = (unsigned short)(w >> 16);
      }
      v16bf b = __builtin_bit_cast(v16bf, bu);

      acc[nt] = __builtin_amdgcn_wmma_f32_16x16x32_bf16(
          false, a, false, b, (short)0, acc[nt], false, false);
    }
  }

  // C/D layout: VGPR r, lanes 0-15 -> M=r, lanes 16-31 -> M=8+r; N = lane&15
  const int mOff = hiHalf ? 8 : 0;
#pragma unroll
  for (int nt = 0; nt < 8; ++nt) {
    int col = nBase + nt * 16 + lan15;
    float bv = bias[col];
#pragma unroll
    for (int i = 0; i < 8; ++i) {
      int row = mWave + i + mOff;
      xg[(size_t)row * G_DIM + col] = acc[nt][i] + bv;
    }
  }
}

// ---------------- device-wide barrier for the persistent scan ----------------
__device__ __forceinline__ void grid_barrier(unsigned* sync) {
  __syncthreads();
  if (threadIdx.x == 0) {
    unsigned* cnt = sync;
    unsigned* gen = sync + 1;
    unsigned g = __hip_atomic_load(gen, __ATOMIC_RELAXED, __HIP_MEMORY_SCOPE_AGENT);
    __threadfence();  // publish this block's gate stores
    unsigned prev = __hip_atomic_fetch_add(cnt, 1u, __ATOMIC_ACQ_REL,
                                           __HIP_MEMORY_SCOPE_AGENT);
    if (prev == NB_SCAN - 1) {
      __hip_atomic_store(cnt, 0u, __ATOMIC_RELAXED, __HIP_MEMORY_SCOPE_AGENT);
      __hip_atomic_store(gen, g + 1, __ATOMIC_RELEASE, __HIP_MEMORY_SCOPE_AGENT);
    } else {
      while (__hip_atomic_load(gen, __ATOMIC_ACQUIRE,
                               __HIP_MEMORY_SCOPE_AGENT) == g)
        __builtin_amdgcn_s_sleep(1);
    }
    __threadfence();
  }
  __syncthreads();
}

// ---------------- persistent LSTM scan ----------------
// 32 blocks x 256 threads = 8192 threads: one gate row each.
// W_hh kept in bf16 (32 MB -> resident in 192 MB L2).
// Each block holds a private (h, c) copy in LDS -> one grid barrier per step,
// gate pre-activations double-buffered in global by step parity.
__global__ __launch_bounds__(256) void lstm_scan_kernel(
    const float* __restrict__ xg,             // [T, G]
    const unsigned short* __restrict__ whh,   // [G, H] bf16
    const float* __restrict__ wout,           // [H]
    const float* __restrict__ bout,           // [1]
    float* __restrict__ out,                  // [T]
    float* __restrict__ gbuf,                 // [2, G] pre-activations
    unsigned* __restrict__ sync)
{
  __shared__ float h_lds[H_DIM];
  __shared__ float c_lds[H_DIM];
  __shared__ float red[256];

  const int tid = threadIdx.x;
  const int r   = blockIdx.x * 256 + tid;     // this thread's gate row

  for (int j = tid; j < H_DIM; j += 256) { h_lds[j] = 0.0f; c_lds[j] = 0.0f; }
  __syncthreads();

  const unsigned short* wrow = whh + (size_t)r * H_DIM;

  for (int t = 0; t < T_LEN; ++t) {
    float* buf = gbuf + (size_t)(t & 1) * G_DIM;

    // gate GEMV row: dot(W_hh[r,:], h) with bf16 weights, fp32 h from LDS
    float acc = 0.0f;
    for (int k = 0; k < H_DIM; k += 8) {
      uint4  wv = *(const uint4*)(wrow + k);
      float4 h0 = *(const float4*)&h_lds[k];
      float4 h1 = *(const float4*)&h_lds[k + 4];
      acc += bf_lo(wv.x) * h0.x + bf_hi(wv.x) * h0.y
           + bf_lo(wv.y) * h0.z + bf_hi(wv.y) * h0.w
           + bf_lo(wv.z) * h1.x + bf_hi(wv.z) * h1.y
           + bf_lo(wv.w) * h1.z + bf_hi(wv.w) * h1.w;
    }
    buf[r] = xg[(size_t)t * G_DIM + r] + acc;

    grid_barrier(sync);

    // pointwise update, redundantly per block (keeps h/c local in LDS)
    for (int j = tid; j < H_DIM; j += 256) {
      float iv = sigmoidf(buf[j]);
      float fv = sigmoidf(buf[H_DIM + j]);
      float gv = tanhf(buf[2 * H_DIM + j]);
      float ov = sigmoidf(buf[3 * H_DIM + j]);
      float cn = fv * c_lds[j] + iv * gv;
      c_lds[j] = cn;
      h_lds[j] = ov * tanhf(cn);
    }
    __syncthreads();

    // output projection: block 0 only
    if (blockIdx.x == 0) {
      float p = 0.0f;
      for (int j = tid; j < H_DIM; j += 256) p += wout[j] * h_lds[j];
      red[tid] = p;
      __syncthreads();
      for (int s = 128; s > 0; s >>= 1) {
        if (tid < s) red[tid] += red[tid + s];
        __syncthreads();
      }
      if (tid == 0) out[t] = red[0] + bout[0];
    }
  }
}

// ---------------- host launcher ----------------
extern "C" void kernel_launch(void* const* d_in, const int* in_sizes, int n_in,
                              void* d_out, int out_size, void* d_ws, size_t ws_size,
                              hipStream_t stream) {
  const float* input = (const float*)d_in[0];   // [T, 1, F]
  const float* W_ih  = (const float*)d_in[1];   // [4H, F]
  const float* W_hh  = (const float*)d_in[2];   // [4H, H]
  const float* b_ih  = (const float*)d_in[3];   // [4H]
  const float* b_hh  = (const float*)d_in[4];   // [4H]
  const float* W_out = (const float*)d_in[5];   // [1, H]
  const float* b_out = (const float*)d_in[6];   // [1]
  float* out = (float*)d_out;                   // [T]

  // workspace layout (bytes)
  char* ws = (char*)d_ws;
  const size_t XG_BYTES   = (size_t)T_LEN * G_DIM * 4;          // 256 MB
  const size_t AB_BYTES   = (size_t)T_LEN * F_DIM * 2;          // 16 MB
  const size_t WIH_BYTES  = (size_t)G_DIM * F_DIM * 2;          // 16 MB
  const size_t WHH_BYTES  = (size_t)G_DIM * H_DIM * 2;          // 32 MB
  const size_t BIAS_BYTES = (size_t)G_DIM * 4;

  float*          xg   = (float*)ws;
  unsigned short* Abf  = (unsigned short*)(ws + XG_BYTES);
  unsigned short* Wihb = (unsigned short*)(ws + XG_BYTES + AB_BYTES);
  unsigned short* Whhb = (unsigned short*)(ws + XG_BYTES + AB_BYTES + WIH_BYTES);
  float*          bias = (float*)(ws + XG_BYTES + AB_BYTES + WIH_BYTES + WHH_BYTES);
  float*          gbuf = (float*)(ws + XG_BYTES + AB_BYTES + WIH_BYTES + WHH_BYTES + BIAS_BYTES);
  unsigned*       sync = (unsigned*)(ws + XG_BYTES + AB_BYTES + WIH_BYTES + WHH_BYTES + BIAS_BYTES
                                     + (size_t)2 * G_DIM * 4);

  cvt_bf16_kernel<<<512, 256, 0, stream>>>(input, Abf,  T_LEN * F_DIM);
  cvt_bf16_kernel<<<512, 256, 0, stream>>>(W_ih,  Wihb, G_DIM * F_DIM);
  cvt_bf16_kernel<<<512, 256, 0, stream>>>(W_hh,  Whhb, G_DIM * H_DIM);
  bias_sum_kernel<<<G_DIM / 256, 256, 0, stream>>>(b_ih, b_hh, bias, G_DIM);
  init_sync_kernel<<<1, 64, 0, stream>>>(sync);

  dim3 g(G_DIM / 128, T_LEN / 128);   // 64 x 64 workgroups
  xgates_wmma_kernel<<<g, 256, 0, stream>>>(Abf, Wihb, bias, xg);

  lstm_scan_kernel<<<NB_SCAN, 256, 0, stream>>>(xg, Whhb, W_out, b_out, out,
                                                gbuf, sync);
}
